// TreeNeighboursGNN_79224966742700
// MI455X (gfx1250) — matrix operand order repaired and checked by reading, not compiled, not run on hardware
//
#include <hip/hip_runtime.h>

#define DEV __device__ __forceinline__

typedef __attribute__((ext_vector_type(16))) __bf16 v16bf;
typedef __attribute__((ext_vector_type(8)))  float  v8f;
typedef __attribute__((ext_vector_type(4)))  float  f4;
typedef __attribute__((ext_vector_type(8)))  unsigned short u16x8;
typedef __attribute__((ext_vector_type(4)))  unsigned short u16x4;

union FragBF { v16bf v; u16x8 h[2]; };

constexpr int  Nn  = 131072;
constexpr int  Ee  = 131072;
constexpr int  Rr  = 4096;
constexpr int  Hd  = 512;
constexpr int  Ll  = 4;
constexpr int  OUTD = 257;
constexpr long LHH = (long)Ll * Hd * Hd;   // elements of one (L,H,H) weight stack

// ---------- bf16 split helpers (RNE, bit ops -> no __bf16 scalar arithmetic) ----------
DEV unsigned short f2bf(float f) {
  unsigned int u = __float_as_uint(f);
  u += 0x7FFFu + ((u >> 16) & 1u);
  return (unsigned short)(u >> 16);
}
DEV float bf2f(unsigned short h) { return __uint_as_float(((unsigned int)h) << 16); }
DEV void splitbf(float f, unsigned short& hi, unsigned short& lo) {
  hi = f2bf(f);
  lo = f2bf(f - bf2f(hi));
}
DEV void atomAddGlobal(float* p, float v) {
  __hip_atomic_fetch_add(p, v, __ATOMIC_RELAXED, __HIP_MEMORY_SCOPE_AGENT);
}
DEV void atomAddShared(float* p, float v) {
  __hip_atomic_fetch_add(p, v, __ATOMIC_RELAXED, __HIP_MEMORY_SCOPE_WORKGROUP);
}
// low 32 bits of a generic pointer into LDS == LDS byte offset (flat aperture rule)
DEV unsigned ldsAddr32(const void* p) { return (unsigned)(uintptr_t)p; }

// ---------- generic zero (f4 granularity) ----------
__global__ void zero_kernel(float* __restrict__ p, long n4) {
  long i = (long)blockIdx.x * blockDim.x + threadIdx.x;
  if (i < n4) {
    f4 z = {0.f, 0.f, 0.f, 0.f};
    *(f4*)(p + i * 4) = z;
  }
}

// ---------- split W1,W2 into bf16 hi/lo once ----------
__global__ void prep_w_kernel(const float* __restrict__ W1, const float* __restrict__ W2,
                              unsigned short* __restrict__ whi, unsigned short* __restrict__ wlo) {
  long i = (long)blockIdx.x * blockDim.x + threadIdx.x;  // over 2*LHH/4
  long e = i * 4;
  const float* src = (e < LHH) ? (W1 + e) : (W2 + (e - LHH));
  f4 v = *(const f4*)src;
  u16x4 hi, lo;
#pragma unroll
  for (int c = 0; c < 4; ++c) {
    unsigned short h, l;
    splitbf(v[c], h, l);
    hi[c] = h; lo[c] = l;
  }
  *(u16x4*)(whi + e) = hi;
  *(u16x4*)(wlo + e) = lo;
}

// ---------- x = key_emb[x_key] + val_emb[x_val] ----------
__global__ void embed_kernel(const int* __restrict__ xk, const int* __restrict__ xv,
                             const float* __restrict__ ke, const float* __restrict__ ve,
                             float* __restrict__ x) {
  long idx = (long)blockIdx.x * blockDim.x + threadIdx.x;  // N*H/4 threads
  int n = (int)(idx >> 7);
  int c = (int)(idx & 127) << 2;
  f4 a = *(const f4*)(ke + (long)xk[n] * Hd + c);
  f4 b = *(const f4*)(ve + (long)xv[n] * Hd + c);
  f4 r = a + b;
  *(f4*)(x + (long)n * Hd + c) = r;
}

// ---------- agg[dst] += x[src] (segment_sum) ----------
__global__ void scatter_kernel(const int* __restrict__ edge, const float* __restrict__ x,
                               float* __restrict__ agg) {
  long idx = (long)blockIdx.x * blockDim.x + threadIdx.x;  // E*H/4 threads
  int e = (int)(idx >> 7);
  int c = (int)(idx & 127) << 2;
  int s = edge[e];
  int d = edge[Ee + e];
  f4 v = *(const f4*)(x + (long)s * Hd + c);
  float* base = agg + (long)d * Hd + c;
  atomAddGlobal(base + 0, v[0]);
  atomAddGlobal(base + 1, v[1]);
  atomAddGlobal(base + 2, v[2]);
  atomAddGlobal(base + 3, v[3]);
}

// ---------- fused GEMM: Y = prologue(A) @ W^T + bias, + column sum/sumsq for BN ----------
// mode 0: A = A0 + A1               (x + agg)
// mode 1: A = relu(S[k]*A0 + T[k])  (BN-apply + ReLU of previous GEMM output)
// bf16x3 fp32 emulation: acc += Ahi*Bhi + Ahi*Blo + Alo*Bhi  (f32 accumulate)
// B tiles staged with GLOBAL_LOAD_ASYNC_TO_LDS_B128 (ASYNCcnt), double-buffered LDS pipeline.
__global__ __launch_bounds__(256)
void gemm_kernel(const float* __restrict__ A0, const float* __restrict__ A1,
                 const float* __restrict__ Svec, const float* __restrict__ Tvec,
                 const unsigned short* __restrict__ Bhi, const unsigned short* __restrict__ Blo,
                 const float* __restrict__ bias, float* __restrict__ Y,
                 float* __restrict__ colsum, float* __restrict__ colsumsq, int mode) {
  const int tid   = threadIdx.x;
  const int lane  = tid & 31;
  const int wave  = tid >> 5;
  const int waveM = wave & 1;   // 2 waves along M
  const int waveN = wave >> 1;  // 4 waves along N
  const int rowBase = blockIdx.y * 128;
  const int colBase = blockIdx.x * 128;

  __shared__ unsigned short AsHi[2][128][40];
  __shared__ unsigned short AsLo[2][128][40];
  __shared__ unsigned short BsHi[2][128][40];
  __shared__ unsigned short BsLo[2][128][40];
  __shared__ float csum[128];
  __shared__ float csq[128];

  v8f acc[4][2];
#pragma unroll
  for (int i = 0; i < 4; ++i)
#pragma unroll
    for (int j = 0; j < 2; ++j)
#pragma unroll
      for (int r = 0; r < 8; ++r) acc[i][j][r] = 0.f;

  const int a_c4 = tid & 7;   // float4 column within 32-wide K slice
  const int a_r  = tid >> 3;  // 0..31
  const int b_c8 = tid & 3;   // u16x8 column
  const int b_r  = tid >> 2;  // 0..63
  const int half = lane >> 4; // K half-select per ISA wave32 layout
  const int mrow = lane & 15;

  // async weight-tile stage: cache/memory -> LDS, no VGPR staging (ASYNCcnt)
  auto stageB = [&](int k0, int buf) {
#pragma unroll
    for (int p = 0; p < 2; ++p) {
      int j = b_r + 64 * p;
      unsigned goff = (unsigned)(((colBase + j) * Hd + k0 + b_c8 * 8) * 2);  // bytes
      unsigned lhi = ldsAddr32(&BsHi[buf][j][b_c8 * 8]);
      unsigned llo = ldsAddr32(&BsLo[buf][j][b_c8 * 8]);
      asm volatile("global_load_async_to_lds_b128 %0, %1, %2"
                   :: "v"(lhi), "v"(goff), "s"(Bhi) : "memory");
      asm volatile("global_load_async_to_lds_b128 %0, %1, %2"
                   :: "v"(llo), "v"(goff), "s"(Blo) : "memory");
    }
  };

  // activation-tile stage: global f32 -> fused prologue -> bf16 hi/lo split -> LDS
  auto stageA = [&](int k0, int buf) {
#pragma unroll
    for (int p = 0; p < 4; ++p) {
      int r = a_r + 32 * p;
      long gi = (long)(rowBase + r) * Hd + k0 + a_c4 * 4;
      f4 v;
      if (mode == 0) {
        f4 u = *(const f4*)(A0 + gi);
        f4 w = *(const f4*)(A1 + gi);
        v = u + w;
      } else {
        f4 u = *(const f4*)(A0 + gi);
        int kc = k0 + a_c4 * 4;
        f4 s = *(const f4*)(Svec + kc);
        f4 t = *(const f4*)(Tvec + kc);
#pragma unroll
        for (int c = 0; c < 4; ++c) v[c] = fmaxf(fmaf(s[c], u[c], t[c]), 0.f);
      }
      u16x4 hi, lo;
#pragma unroll
      for (int c = 0; c < 4; ++c) {
        unsigned short h, l;
        splitbf(v[c], h, l);
        hi[c] = h; lo[c] = l;
      }
      *(u16x4*)&AsHi[buf][r][a_c4 * 4] = hi;
      *(u16x4*)&AsLo[buf][r][a_c4 * 4] = lo;
    }
  };

  auto computeTile = [&](int buf) {
    FragBF aHi[4], aLo[4], bHi[2], bLo[2];
#pragma unroll
    for (int tm = 0; tm < 4; ++tm) {
      int r = waveM * 64 + tm * 16 + mrow;
      int c0 = half * 8;  // lanes<16: K0..7 & K16..23 ; lanes>=16: K8..15 & K24..31
      aHi[tm].h[0] = *(const u16x8*)&AsHi[buf][r][c0];
      aHi[tm].h[1] = *(const u16x8*)&AsHi[buf][r][c0 + 16];
      aLo[tm].h[0] = *(const u16x8*)&AsLo[buf][r][c0];
      aLo[tm].h[1] = *(const u16x8*)&AsLo[buf][r][c0 + 16];
    }
#pragma unroll
    for (int tn = 0; tn < 2; ++tn) {
      int j = waveN * 32 + tn * 16 + mrow;
      int kk = half * 8;  // lanes<16: K0..7 & K16..23 ; lanes>=16: K8..15 & K24..31
      bHi[tn].h[0] = *(const u16x8*)&BsHi[buf][j][kk];
      bHi[tn].h[1] = *(const u16x8*)&BsHi[buf][j][kk + 16];
      bLo[tn].h[0] = *(const u16x8*)&BsLo[buf][j][kk];
      bLo[tn].h[1] = *(const u16x8*)&BsLo[buf][j][kk + 16];
    }
#pragma unroll
    for (int tm = 0; tm < 4; ++tm)
#pragma unroll
      for (int tn = 0; tn < 2; ++tn) {
        v8f c = acc[tm][tn];
        c = __builtin_amdgcn_wmma_f32_16x16x32_bf16(false, aLo[tm].v, false, bHi[tn].v,
                                                    (short)0, c, false, false);
        c = __builtin_amdgcn_wmma_f32_16x16x32_bf16(false, aHi[tm].v, false, bLo[tn].v,
                                                    (short)0, c, false, false);
        c = __builtin_amdgcn_wmma_f32_16x16x32_bf16(false, aHi[tm].v, false, bHi[tn].v,
                                                    (short)0, c, false, false);
        acc[tm][tn] = c;
      }
  };

  // software pipeline: stage(it+1) overlaps compute(it); one barrier per iteration.
  stageB(0, 0);
  stageA(0, 0);
  for (int it = 0; it < 16; ++it) {
    int cur = it & 1;
    asm volatile("s_wait_asynccnt 0" ::: "memory");  // my async LDS writes visible
    __syncthreads();                                 // everyone staged buf `cur`
    if (it < 15) {
      stageB((it + 1) * 32, cur ^ 1);
      stageA((it + 1) * 32, cur ^ 1);
    }
    computeTile(cur);
  }

  // epilogue: bias add, store, per-column BN partial stats
  if (tid < 128) { csum[tid] = 0.f; csq[tid] = 0.f; }
  __syncthreads();

#pragma unroll
  for (int tn = 0; tn < 2; ++tn) {
    int nl = waveN * 32 + tn * 16 + mrow;
    int gcol = colBase + nl;
    float bj = bias[gcol];
    float ps = 0.f, pq = 0.f;
#pragma unroll
    for (int tm = 0; tm < 4; ++tm) {
#pragma unroll
      for (int r = 0; r < 8; ++r) {
        int ml = waveM * 64 + tm * 16 + half * 8 + r;  // C/D layout: lanes>=16 -> M+8
        float y = acc[tm][tn][r] + bj;
        Y[(long)(rowBase + ml) * Hd + gcol] = y;
        ps += y;
        pq += y * y;
      }
    }
    atomAddShared(&csum[nl], ps);
    atomAddShared(&csq[nl], pq);
  }
  __syncthreads();
  if (tid < 128) {
    atomAddGlobal(&colsum[colBase + tid], csum[tid]);
    atomAddGlobal(&colsumsq[colBase + tid], csq[tid]);
  }
}

// ---------- BN stats -> per-column scale/shift: y = s*x + t ----------
__global__ void bnstats_kernel(const float* __restrict__ colsum, const float* __restrict__ colsumsq,
                               const float* __restrict__ g, const float* __restrict__ be,
                               float* __restrict__ s, float* __restrict__ t) {
  int j = threadIdx.x;  // 512
  const float inv = 1.0f / (float)Nn;
  float m  = colsum[j] * inv;
  float var = colsumsq[j] * inv - m * m;
  float rs = rsqrtf(var + 1e-5f);
  float sc = g[j] * rs;
  s[j] = sc;
  t[j] = fmaf(-m, sc, be[j]);
}

// ---------- residual + LayerNorm: x = LN(x + relu(s*z2 + t)) ----------
__global__ __launch_bounds__(256)
void post_kernel(float* __restrict__ x, const float* __restrict__ z2,
                 const float* __restrict__ s, const float* __restrict__ t,
                 const float* __restrict__ lg, const float* __restrict__ lb) {
  int row = blockIdx.x;
  int tid = threadIdx.x;
  __shared__ float red[256];
  __shared__ float red2[256];
  const float* zr = z2 + (long)row * Hd;
  float* xr = x + (long)row * Hd;
  int c0 = tid, c1 = tid + 256;
  float v0 = xr[c0] + fmaxf(fmaf(s[c0], zr[c0], t[c0]), 0.f);
  float v1 = xr[c1] + fmaxf(fmaf(s[c1], zr[c1], t[c1]), 0.f);
  red[tid]  = v0 + v1;
  red2[tid] = v0 * v0 + v1 * v1;
  __syncthreads();
  for (int st = 128; st > 0; st >>= 1) {
    if (tid < st) { red[tid] += red[tid + st]; red2[tid] += red2[tid + st]; }
    __syncthreads();
  }
  float m  = red[0] * (1.f / 512.f);
  float var = red2[0] * (1.f / 512.f) - m * m;
  float rs = rsqrtf(var + 1e-5f);
  xr[c0] = (v0 - m) * rs * lg[c0] + lb[c0];
  xr[c1] = (v1 - m) * rs * lg[c1] + lb[c1];
}

// ---------- out = x[roots] @ Wout^T  (16 roots per block, Wout streamed once/block) ----------
__global__ __launch_bounds__(256)
void out_kernel(const float* __restrict__ x, const int* __restrict__ roots,
                const float* __restrict__ Wout, float* __restrict__ out) {
  int blk = blockIdx.x;  // Rr/16 blocks
  int tid = threadIdx.x;
  __shared__ float xs[16][512];
#pragma unroll
  for (int p = 0; p < 8; ++p) {
    int idx = tid + 256 * p;
    int rr = idx >> 7;
    int c = (idx & 127) << 2;
    int node = roots[blk * 16 + rr];
    *(f4*)&xs[rr][c] = *(const f4*)(x + (long)node * Hd + c);
  }
  __syncthreads();
  for (int o = tid; o < OUTD; o += 256) {
    float accv[16];
#pragma unroll
    for (int r = 0; r < 16; ++r) accv[r] = 0.f;
    const float* wrow = Wout + (long)o * Hd;
    for (int k = 0; k < Hd; k += 4) {
      f4 w = *(const f4*)(wrow + k);
#pragma unroll
      for (int r = 0; r < 16; ++r) {
        f4 xv = *(const f4*)&xs[r][k];
        accv[r] = fmaf(w[0], xv[0], fmaf(w[1], xv[1], fmaf(w[2], xv[2], fmaf(w[3], xv[3], accv[r]))));
      }
    }
#pragma unroll
    for (int r = 0; r < 16; ++r)
      out[(long)(blk * 16 + r) * OUTD + o] = accv[r];
  }
}

extern "C" void kernel_launch(void* const* d_in, const int* in_sizes, int n_in,
                              void* d_out, int out_size, void* d_ws, size_t ws_size,
                              hipStream_t stream) {
  const int*   x_key = (const int*)d_in[0];
  const int*   x_val = (const int*)d_in[1];
  const int*   edge  = (const int*)d_in[2];
  const int*   roots = (const int*)d_in[3];
  const float* ke    = (const float*)d_in[4];
  const float* ve    = (const float*)d_in[5];
  const float* W1    = (const float*)d_in[6];
  const float* b1    = (const float*)d_in[7];
  const float* g1    = (const float*)d_in[8];
  const float* be1   = (const float*)d_in[9];
  const float* W2    = (const float*)d_in[10];
  const float* b2    = (const float*)d_in[11];
  const float* g2    = (const float*)d_in[12];
  const float* be2   = (const float*)d_in[13];
  const float* ln_g  = (const float*)d_in[14];
  const float* ln_b  = (const float*)d_in[15];
  const float* Wout  = (const float*)d_in[16];

  char* ws = (char*)d_ws;
  const size_t XB = (size_t)Nn * Hd * sizeof(float);  // 256 MB
  float* x  = (float*)(ws);
  float* z1 = (float*)(ws + XB);
  float* z2 = (float*)(ws + 2 * XB);                  // doubles as agg
  unsigned short* whi = (unsigned short*)(ws + 3 * XB);
  unsigned short* wlo = whi + 2 * LHH;
  float* colsum   = (float*)(ws + 3 * XB + 4 * LHH * sizeof(unsigned short));
  float* colsumsq = colsum + Hd;
  float* sbuf     = colsum + 2 * Hd;
  float* tbuf     = colsum + 3 * Hd;

  const int NH4 = Nn * Hd / 4;        // 16,777,216 f4
  const int EH4 = Ee * Hd / 4;

  // 1) pre-split weights to bf16 hi/lo
  prep_w_kernel<<<(int)(2 * LHH / 4 / 256), 256, 0, stream>>>(W1, W2, whi, wlo);
  // 2) embeddings
  embed_kernel<<<NH4 / 256, 256, 0, stream>>>(x_key, x_val, ke, ve, x);

  for (int i = 0; i < Ll; ++i) {
    float* agg = z2;
    zero_kernel<<<NH4 / 256, 256, 0, stream>>>(agg, NH4);
    scatter_kernel<<<EH4 / 256, 256, 0, stream>>>(edge, x, agg);

    // GEMM1: z1 = (x + agg) @ W1[i]^T + b1[i], collect column stats
    zero_kernel<<<1, 256, 0, stream>>>(colsum, 256);  // 1024 floats = 256 f4
    gemm_kernel<<<dim3(Hd / 128, Nn / 128), 256, 0, stream>>>(
        x, agg, nullptr, nullptr,
        whi + (size_t)i * Hd * Hd, wlo + (size_t)i * Hd * Hd,
        b1 + i * Hd, z1, colsum, colsumsq, 0);
    bnstats_kernel<<<1, Hd, 0, stream>>>(colsum, colsumsq, g1 + i * Hd, be1 + i * Hd, sbuf, tbuf);

    // GEMM2: z2 = relu(bn1(z1)) @ W2[i]^T + b2[i], collect column stats
    zero_kernel<<<1, 256, 0, stream>>>(colsum, 256);
    gemm_kernel<<<dim3(Hd / 128, Nn / 128), 256, 0, stream>>>(
        z1, nullptr, sbuf, tbuf,
        whi + LHH + (size_t)i * Hd * Hd, wlo + LHH + (size_t)i * Hd * Hd,
        b2 + i * Hd, z2, colsum, colsumsq, 1);
    bnstats_kernel<<<1, Hd, 0, stream>>>(colsum, colsumsq, g2 + i * Hd, be2 + i * Hd, sbuf, tbuf);

    // x = LayerNorm(x + relu(bn2(z2)))
    post_kernel<<<Nn, 256, 0, stream>>>(x, z2, sbuf, tbuf, ln_g + i * Hd, ln_b + i * Hd);
  }

  // output: out = x[roots] @ Wout^T
  out_kernel<<<Rr / 16, 256, 0, stream>>>(x, roots, Wout, (float*)d_out);
}
